// HGSA_v13_40475771797634
// MI455X (gfx1250) — compile-verified, matrix operand order checked by reading, not run on hardware
//
#include <hip/hip_runtime.h>
#include <hip/hip_bf16.h>
#include <math.h>

typedef __attribute__((ext_vector_type(16))) _Float16 v16h;
typedef __attribute__((ext_vector_type(8)))  float    v8f;

#define NBLK   3
#define CDIM   64
#define NHEADS 4
#define DHEAD  16
#define BATCH  2
#define HIMG   256
#define WIMG   256
#define HWPIX  (HIMG*WIMG)          // 65536 pixels per batch
#define NROWS  (BATCH*HWPIX)        // 131072 rows total
#define NSPLIT 256                  // split-K factor for Gram reduction
#define FFC    256                  // FFN hidden width

__device__ __forceinline__ float gelu_f(float x) {
  return 0.5f * x * (1.0f + erff(x * 0.70710678118654752f));
}

// Inverse of the WMMA 16-bit K-striping: given k (0..31) within a tile,
// return (half, j) such that lane-group `half` holds k at element pair j.
//   kk(j, half) = j<4 ? 8*half + 2j : 16 + 8*half + 2(j-4)
__device__ __forceinline__ void kswz(int k, int& half, int& elem) {
  int kpair = k >> 1;
  half = (kpair >> 2) & 1;
  int j = (kpair & 3) + ((kpair >> 3) << 2);
  elem = 2 * j + (k & 1);
}

// ---------------------------------------------------------------------------
// Layout transforms
// ---------------------------------------------------------------------------
__global__ __launch_bounds__(256) void k_nchw2nhwc_f32(const float* __restrict__ in,
                                                       float* __restrict__ out) {
  long long idx = (long long)blockIdx.x * 256 + threadIdx.x;
  if (idx >= (long long)NROWS * CDIM) return;
  int c = (int)(idx & 63);
  long long p = idx >> 6;
  int b = (int)(p / HWPIX), pix = (int)(p % HWPIX);
  out[idx] = in[((size_t)b * CDIM + c) * HWPIX + pix];
}

__global__ __launch_bounds__(256) void k_nchw2nhwc_f16(const float* __restrict__ in,
                                                       _Float16* __restrict__ out) {
  long long idx = (long long)blockIdx.x * 256 + threadIdx.x;
  if (idx >= (long long)NROWS * CDIM) return;
  int c = (int)(idx & 63);
  long long p = idx >> 6;
  int b = (int)(p / HWPIX), pix = (int)(p % HWPIX);
  out[idx] = (_Float16)in[((size_t)b * CDIM + c) * HWPIX + pix];
}

__global__ __launch_bounds__(256) void k_nhwc2nchw_f32(const float* __restrict__ in,
                                                       float* __restrict__ out) {
  long long idx = (long long)blockIdx.x * 256 + threadIdx.x;
  if (idx >= (long long)NROWS * CDIM) return;
  int pix = (int)(idx % HWPIX);
  int c   = (int)((idx / HWPIX) % CDIM);
  int b   = (int)(idx / ((long long)HWPIX * CDIM));
  out[idx] = in[((size_t)b * HWPIX + pix) * CDIM + c];
}

// ---------------------------------------------------------------------------
// WMMA GEMM: Out[N, COUT] = epilogue( A[N, CIN] @ W[CIN, COUT] )
// 256 threads = 8 waves -> 64(M) x 32(N) block tile; wave = one 16x16 WMMA,
// K stepped by 32. Tiles live in LDS in *fragment layout*: each lane's whole
// 16-half fragment is a contiguous 32B run -> two ds_load_b128 per operand.
// For f16 A inputs the A tile is staged with GLOBAL_LOAD_ASYNC_TO_LDS_B128
// (one 16B chunk per thread; chunk q of row r lands contiguously at lane
// (q&1)*16 + (r&15), elem (q>>1)*8), synchronized with s_wait_asynccnt.
// ---------------------------------------------------------------------------
template<int CIN, int COUT, bool AF32, bool HASB, bool GEL,
         bool ADD32, bool ADD16, bool OUT32, bool WPB>
__global__ __launch_bounds__(256) void k_gemm(
    const void* __restrict__ Ap, const float* __restrict__ W,
    const float* __restrict__ bias, const float* __restrict__ res32,
    const _Float16* __restrict__ res16, void* __restrict__ Outp,
    int rowsPerBatch)
{
  __shared__ __attribute__((aligned(32))) _Float16 AsF[4][32][16]; // [m-tile][lane][frag]
  __shared__ __attribute__((aligned(32))) _Float16 BsF[2][32][16]; // [n-tile][lane][frag]
  const int tid  = threadIdx.x;
  const int row0 = blockIdx.x * 64;
  const int col0 = blockIdx.y * 32;
  const int wave = tid >> 5;
  const int lane = tid & 31;
  const int wm   = wave & 3;   // 4 row-tiles
  const int wn   = wave >> 2;  // 2 col-tiles

  const float* Wb = W;
  if (WPB) Wb = W + (size_t)(row0 / rowsPerBatch) * CIN * COUT;

  // precomputed for async A staging: this thread owns (row r, 16B chunk q)
  const int ar = tid >> 2;
  const int aq = tid & 3;
  _Float16* aldsp = &AsF[ar >> 4][(aq & 1) * 16 + (ar & 15)][(aq >> 1) * 8];
  const unsigned aldsOff = (unsigned)(size_t)(void*)aldsp; // low 32b = LDS offset

  v8f acc = {};
  for (int kt = 0; kt < CIN / 32; ++kt) {
    // ---- stage A tile (64 rows x 32 k) into fragment layout ----
    if constexpr (AF32) {
      #pragma unroll
      for (int t = 0; t < 8; ++t) {
        int e = tid + t * 256;
        int r = e >> 5, c = e & 31;
        float v = ((const float*)Ap)[(size_t)(row0 + r) * CIN + kt * 32 + c];
        int half, el; kswz(c, half, el);
        AsF[r >> 4][half * 16 + (r & 15)][el] = (_Float16)v;
      }
    } else {
      const _Float16* gp = (const _Float16*)Ap +
                           ((size_t)(row0 + ar) * CIN + kt * 32 + aq * 8);
      unsigned long long ga = (unsigned long long)(size_t)gp;
      asm volatile("global_load_async_to_lds_b128 %0, %1, off"
                   :: "v"(aldsOff), "v"(ga) : "memory");
    }
    // ---- stage B tile (32 k x 32 n), fp32 weights -> f16 fragments ----
    #pragma unroll
    for (int t = 0; t < 4; ++t) {
      int e = tid + t * 256;
      int k = e >> 5, n = e & 31;
      float v = Wb[(size_t)(kt * 32 + k) * COUT + col0 + n];
      int half, el; kswz(k, half, el);
      BsF[n >> 4][half * 16 + (n & 15)][el] = (_Float16)v;
    }
    if constexpr (!AF32) {
      asm volatile("s_wait_asynccnt 0x0" ::: "memory");
    }
    __syncthreads();

    v16h a   = *(const v16h*)(&AsF[wm][lane][0]);
    v16h bfr = *(const v16h*)(&BsF[wn][lane][0]);
    acc = __builtin_amdgcn_wmma_f32_16x16x32_f16(false, a, false, bfr,
                                                 (short)0, acc, false, false);
    __syncthreads();
  }

  // epilogue: C/D layout -> lane = N, VGPR r = M (M+8 for lanes 16-31)
  const int gn  = col0 + wn * 16 + (lane & 15);
  const int mhi = (lane >> 4) * 8;
  #pragma unroll
  for (int r = 0; r < 8; ++r) {
    int gm = row0 + wm * 16 + mhi + r;
    float v = acc[r];
    if constexpr (HASB)  v += bias[gn];
    if constexpr (ADD16) v += (float)res16[(size_t)gm * COUT + gn];
    if constexpr (ADD32) v += res32[(size_t)gm * COUT + gn];
    if constexpr (GEL)   v = gelu_f(v);
    if constexpr (OUT32) ((float*)Outp)[(size_t)gm * COUT + gn] = v;
    else                 ((_Float16*)Outp)[(size_t)gm * COUT + gn] = (_Float16)v;
  }
}

// ---------------------------------------------------------------------------
// Split-K Gram: partial[bh,s,16,16] = sum_n K[n, h*16+d] * Q[n, h*16+e]
// one wave per workgroup, direct global gather into WMMA fragments
// ---------------------------------------------------------------------------
__global__ __launch_bounds__(32) void k_gram(const _Float16* __restrict__ Kb,
                                             const _Float16* __restrict__ Qb,
                                             float* __restrict__ partial)
{
  const int bh = blockIdx.x;            // b*NHEADS + h
  const int b  = bh / NHEADS;
  const int h  = bh % NHEADS;
  const int s  = blockIdx.y;
  const int chunk = HWPIX / NSPLIT;     // 256
  const int nbase0 = b * HWPIX + s * chunk;
  const int lane = threadIdx.x;
  const int kb = (lane >> 4) * 8;
  const int ch = h * 16 + (lane & 15);

  v8f acc = {};
  for (int t = 0; t < chunk / 32; ++t) {
    int nb = nbase0 + t * 32;
    v16h a, bfr;
    #pragma unroll
    for (int j = 0; j < 8; ++j) {
      int kk = (j < 4) ? (kb + 2 * j) : (16 + kb + 2 * (j - 4));
      a[2 * j]       = Kb[(size_t)(nb + kk) * CDIM + ch];       // A[d, n] = K[n, d]
      a[2 * j + 1]   = Kb[(size_t)(nb + kk + 1) * CDIM + ch];
      bfr[2 * j]     = Qb[(size_t)(nb + kk) * CDIM + ch];       // B[n, e] = Q[n, e]
      bfr[2 * j + 1] = Qb[(size_t)(nb + kk + 1) * CDIM + ch];
    }
    acc = __builtin_amdgcn_wmma_f32_16x16x32_f16(false, a, false, bfr,
                                                 (short)0, acc, false, false);
  }
  float* pp = partial + ((size_t)bh * NSPLIT + s) * 256;
  const int nn  = lane & 15;
  const int mhi = (lane >> 4) * 8;
  #pragma unroll
  for (int r = 0; r < 8; ++r) pp[(mhi + r) * 16 + nn] = acc[r];
}

// ---------------------------------------------------------------------------
// Per-channel L2 norms over n for Q (which=0) and K (which=1)
// ---------------------------------------------------------------------------
__global__ __launch_bounds__(256) void k_norm(const _Float16* __restrict__ Qb,
                                              const _Float16* __restrict__ Kb,
                                              float* __restrict__ norms)
{
  __shared__ float red[256];
  const int id = blockIdx.x;                       // [2][B][64]
  const int which = id / (BATCH * CDIM);
  const int rem = id % (BATCH * CDIM);
  const int b = rem / CDIM, c = rem % CDIM;
  const _Float16* src = which ? Kb : Qb;
  float s = 0.f;
  for (int n = threadIdx.x; n < HWPIX; n += 256) {
    float v = (float)src[((size_t)b * HWPIX + n) * CDIM + c];
    s += v * v;
  }
  red[threadIdx.x] = s;
  __syncthreads();
  for (int o = 128; o > 0; o >>= 1) {
    if (threadIdx.x < o) red[threadIdx.x] += red[threadIdx.x + o];
    __syncthreads();
  }
  if (threadIdx.x == 0) norms[id] = fmaxf(sqrtf(red[0]), 1e-6f);
}

// ---------------------------------------------------------------------------
// attn finalize: reduce partials, normalize, softmax over e, fold into Wp:
//   Wcomb[b][h*16+e][cout] = sum_d attn[h][d][e] * Wp[h*16+d][cout]
// ---------------------------------------------------------------------------
__global__ __launch_bounds__(256) void k_attn_finalize(
    const float* __restrict__ partial, const float* __restrict__ norms,
    const float* __restrict__ rescale, const float* __restrict__ Wp,
    float* __restrict__ Wcomb)
{
  __shared__ float attn[NHEADS][16][16];
  const int b = blockIdx.x;
  for (int idx = threadIdx.x; idx < NHEADS * 256; idx += 256) {
    int h = idx >> 8, de = idx & 255, d = de >> 4, e = de & 15;
    const float* p = partial + ((size_t)(b * NHEADS + h) * NSPLIT) * 256 + de;
    float g = 0.f;
    for (int s = 0; s < NSPLIT; ++s) g += p[(size_t)s * 256];
    float qn = norms[b * CDIM + h * 16 + e];
    float kn = norms[BATCH * CDIM + b * CDIM + h * 16 + d];
    attn[h][d][e] = g / (kn * qn) * rescale[h];
  }
  __syncthreads();
  if (threadIdx.x < NHEADS * 16) {
    int h = threadIdx.x >> 4, d = threadIdx.x & 15;
    float mx = -3.4e38f;
    for (int e = 0; e < 16; ++e) mx = fmaxf(mx, attn[h][d][e]);
    float sum = 0.f;
    float ex[16];
    for (int e = 0; e < 16; ++e) { ex[e] = expf(attn[h][d][e] - mx); sum += ex[e]; }
    float inv = 1.f / sum;
    for (int e = 0; e < 16; ++e) attn[h][d][e] = ex[e] * inv;
  }
  __syncthreads();
  for (int idx = threadIdx.x; idx < CDIM * CDIM; idx += 256) {
    int ein = idx >> 6, cout = idx & 63;
    int h = ein >> 4, e = ein & 15;
    float s = 0.f;
    for (int d = 0; d < 16; ++d)
      s += attn[h][d][e] * Wp[(h * 16 + d) * CDIM + cout];
    Wcomb[(size_t)b * CDIM * CDIM + ein * CDIM + cout] = s;
  }
}

// ---------------------------------------------------------------------------
// 5x5 depthwise conv + sigmoid gate:  vg = v * (m1*sigmoid(dw5(t2)+db) + m1)
// ---------------------------------------------------------------------------
__global__ __launch_bounds__(256) void k_dw5_gate(
    const _Float16* __restrict__ t2, const _Float16* __restrict__ m1,
    const _Float16* __restrict__ v, const float* __restrict__ kern,
    const float* __restrict__ db, _Float16* __restrict__ vg)
{
  long long idx = (long long)blockIdx.x * 256 + threadIdx.x;
  if (idx >= (long long)NROWS * CDIM) return;
  int c = (int)(idx & 63);
  long long p = idx >> 6;
  int pix = (int)(p % HWPIX), b = (int)(p / HWPIX);
  int y = pix >> 8, x = pix & 255;
  float s = 0.f;
  #pragma unroll
  for (int ky = 0; ky < 5; ++ky) {
    int yy = y + ky - 2;
    if ((unsigned)yy >= HIMG) continue;
    #pragma unroll
    for (int kx = 0; kx < 5; ++kx) {
      int xx = x + kx - 2;
      if ((unsigned)xx >= WIMG) continue;
      s += (float)t2[((size_t)b * HWPIX + yy * WIMG + xx) * CDIM + c]
           * kern[(ky * 5 + kx) * CDIM + c];
    }
  }
  float am = 1.f / (1.f + expf(-(s + db[c])));
  float m  = (float)m1[idx];
  float ma = m * am + m;
  vg[idx] = (_Float16)((float)v[idx] * ma);
}

// ---------------------------------------------------------------------------
// 3x3 depthwise conv, optional GELU
// ---------------------------------------------------------------------------
template<int CCH, bool GEL>
__global__ __launch_bounds__(256) void k_dw3(const _Float16* __restrict__ in,
                                             const float* __restrict__ kern,
                                             _Float16* __restrict__ out)
{
  long long idx = (long long)blockIdx.x * 256 + threadIdx.x;
  if (idx >= (long long)NROWS * CCH) return;
  int c = (int)(idx % CCH);
  long long p = idx / CCH;
  int pix = (int)(p % HWPIX), b = (int)(p / HWPIX);
  int y = pix >> 8, x = pix & 255;
  float s = 0.f;
  #pragma unroll
  for (int ky = 0; ky < 3; ++ky) {
    int yy = y + ky - 1;
    if ((unsigned)yy >= HIMG) continue;
    #pragma unroll
    for (int kx = 0; kx < 3; ++kx) {
      int xx = x + kx - 1;
      if ((unsigned)xx >= WIMG) continue;
      s += (float)in[((size_t)b * HWPIX + yy * WIMG + xx) * CCH + c]
           * kern[(ky * 3 + kx) * CCH + c];
    }
  }
  out[idx] = (_Float16)(GEL ? gelu_f(s) : s);
}

// ---------------------------------------------------------------------------
// LayerNorm over last dim (64): one wave per row, wave32 shuffle reduction
// ---------------------------------------------------------------------------
__global__ __launch_bounds__(256) void k_ln(const float* __restrict__ x,
                                            const float* __restrict__ g,
                                            const float* __restrict__ bta,
                                            _Float16* __restrict__ y)
{
  int row  = blockIdx.x * 8 + (threadIdx.x >> 5);
  int lane = threadIdx.x & 31;
  const float* xr = x + (size_t)row * CDIM;
  float v0 = xr[lane], v1 = xr[lane + 32];
  float s = v0 + v1;
  #pragma unroll
  for (int o = 16; o; o >>= 1) s += __shfl_xor(s, o, 32);
  float mu = s * (1.f / 64.f);
  float d0 = v0 - mu, d1 = v1 - mu;
  float q = d0 * d0 + d1 * d1;
  #pragma unroll
  for (int o = 16; o; o >>= 1) q += __shfl_xor(q, o, 32);
  float inv = rsqrtf(q * (1.f / 64.f) + 1e-5f);
  _Float16* yr = y + (size_t)row * CDIM;
  yr[lane]      = (_Float16)(d0 * inv * g[lane] + bta[lane]);
  yr[lane + 32] = (_Float16)(d1 * inv * g[lane + 32] + bta[lane + 32]);
}

// ---------------------------------------------------------------------------
// host launcher
// ---------------------------------------------------------------------------
extern "C" void kernel_launch(void* const* d_in, const int* in_sizes, int n_in,
                              void* d_out, int out_size, void* d_ws, size_t ws_size,
                              hipStream_t stream) {
  (void)in_sizes; (void)n_in; (void)out_size; (void)ws_size;
  const float* x       = (const float*)d_in[0];
  const float* mask    = (const float*)d_in[1];
  const float* Wq      = (const float*)d_in[2];
  const float* Wk      = (const float*)d_in[3];
  const float* Wv      = (const float*)d_in[4];
  const float* rescale = (const float*)d_in[5];
  const float* Wp      = (const float*)d_in[6];
  const float* bp      = (const float*)d_in[7];
  const float* pe1     = (const float*)d_in[8];
  const float* pe2     = (const float*)d_in[9];
  const float* mm_w1   = (const float*)d_in[10];
  const float* mm_b1   = (const float*)d_in[11];
  const float* mm_w2   = (const float*)d_in[12];
  const float* mm_b2   = (const float*)d_in[13];
  const float* mm_dw   = (const float*)d_in[14];
  const float* mm_db   = (const float*)d_in[15];
  const float* ln_g    = (const float*)d_in[16];
  const float* ln_b    = (const float*)d_in[17];
  const float* ff_w1   = (const float*)d_in[18];
  const float* ff_dw   = (const float*)d_in[19];
  const float* ff_w2   = (const float*)d_in[20];

  char* ws = (char*)d_ws;
  size_t off = 0;
  auto alloc = [&](size_t bytes) {
    void* p = ws + off;
    off += (bytes + 255) & ~(size_t)255;
    return p;
  };
  float*    X0 = (float*)   alloc((size_t)NROWS * CDIM * 4);
  float*    X1 = (float*)   alloc((size_t)NROWS * CDIM * 4);
  _Float16* Mh = (_Float16*)alloc((size_t)NROWS * CDIM * 2);
  _Float16* Qb = (_Float16*)alloc((size_t)NROWS * CDIM * 2);
  _Float16* Kb = (_Float16*)alloc((size_t)NROWS * CDIM * 2);
  _Float16* Vb = (_Float16*)alloc((size_t)NROWS * CDIM * 2);
  _Float16* T1 = (_Float16*)alloc((size_t)NROWS * CDIM * 2);
  _Float16* T2 = (_Float16*)alloc((size_t)NROWS * CDIM * 2);
  _Float16* VG = (_Float16*)alloc((size_t)NROWS * CDIM * 2);
  _Float16* H1 = (_Float16*)alloc((size_t)NROWS * FFC * 2);
  _Float16* H2 = (_Float16*)alloc((size_t)NROWS * FFC * 2);
  float* partial = (float*)alloc((size_t)BATCH * NHEADS * NSPLIT * 256 * 4);
  float* norms   = (float*)alloc((size_t)2 * BATCH * CDIM * 4);
  float* Wcomb   = (float*)alloc((size_t)BATCH * CDIM * CDIM * 4);

  const dim3 blk(256);
  const int  gC = (int)(((long long)NROWS * CDIM + 255) / 256);
  const int  gF = (int)(((long long)NROWS * FFC + 255) / 256);
  const dim3 g64(NROWS / 64, CDIM / 32);   // 2048 x 2
  const dim3 gFF(NROWS / 64, FFC / 32);    // 2048 x 8

  k_nchw2nhwc_f32<<<gC, blk, 0, stream>>>(x, X0);
  k_nchw2nhwc_f16<<<gC, blk, 0, stream>>>(mask, Mh);

  for (int i = 0; i < NBLK; ++i) {
    const float* wq = Wq + (size_t)i * CDIM * CDIM;
    const float* wk = Wk + (size_t)i * CDIM * CDIM;
    const float* wv = Wv + (size_t)i * CDIM * CDIM;
    // Q/K/V projections (A = fp32 activations)
    k_gemm<64, 64, true, false, false, false, false, false, false>
        <<<g64, blk, 0, stream>>>(X0, wq, nullptr, nullptr, nullptr, Qb, HWPIX);
    k_gemm<64, 64, true, false, false, false, false, false, false>
        <<<g64, blk, 0, stream>>>(X0, wk, nullptr, nullptr, nullptr, Kb, HWPIX);
    k_gemm<64, 64, true, false, false, false, false, false, false>
        <<<g64, blk, 0, stream>>>(X0, wv, nullptr, nullptr, nullptr, Vb, HWPIX);
    // mask branch: m1 = Mh@w1+b1 ; t2 = m1@w2+b2   (async A staging)
    k_gemm<64, 64, false, true, false, false, false, false, false>
        <<<g64, blk, 0, stream>>>(Mh, mm_w1 + (size_t)i * 4096, mm_b1 + i * 64,
                                  nullptr, nullptr, T1, HWPIX);
    k_gemm<64, 64, false, true, false, false, false, false, false>
        <<<g64, blk, 0, stream>>>(T1, mm_w2 + (size_t)i * 4096, mm_b2 + i * 64,
                                  nullptr, nullptr, T2, HWPIX);
    // gate: vg = v * (m1*sigmoid(dw5(t2)+db) + m1)
    k_dw5_gate<<<gC, blk, 0, stream>>>(T2, T1, Vb, mm_dw + (size_t)i * 25 * 64,
                                       mm_db + i * 64, VG);
    // channel attention: norms, split-K Gram, softmax fold into Wp
    k_norm<<<dim3(2 * BATCH * CDIM), blk, 0, stream>>>(Qb, Kb, norms);
    k_gram<<<dim3(BATCH * NHEADS, NSPLIT), dim3(32), 0, stream>>>(Kb, Qb, partial);
    k_attn_finalize<<<dim3(BATCH), blk, 0, stream>>>(
        partial, norms, rescale + i * NHEADS, Wp + (size_t)i * 4096, Wcomb);
    // positional branch: outp = dw3(gelu(dw3(v, pe1)), pe2)
    k_dw3<64, true><<<gC, blk, 0, stream>>>(Vb, pe1 + (size_t)i * 9 * 64, T1);
    k_dw3<64, false><<<gC, blk, 0, stream>>>(T1, pe2 + (size_t)i * 9 * 64, T2);
    // X1 = vg @ Wcomb[b] + bp + outp + X0   (attention apply + proj + residuals)
    k_gemm<64, 64, false, true, false, true, true, true, true>
        <<<g64, blk, 0, stream>>>(VG, Wcomb, bp + i * 64, X0, T2, X1, HWPIX);
    // FFN: y = LN(X1); H1 = gelu(y@w1); H2 = gelu(dw3(H1)); X0 = H2@w2 + X1
    k_ln<<<dim3(NROWS / 8), blk, 0, stream>>>(X1, ln_g + i * 64, ln_b + i * 64, Qb);
    k_gemm<64, 256, false, false, true, false, false, false, false>
        <<<gFF, blk, 0, stream>>>(Qb, ff_w1 + (size_t)i * 64 * 256, nullptr,
                                  nullptr, nullptr, H1, HWPIX);
    k_dw3<256, true><<<gF, blk, 0, stream>>>(H1, ff_dw + (size_t)i * 9 * 256, H2);
    k_gemm<256, 64, false, false, false, true, false, true, false>
        <<<g64, blk, 0, stream>>>(H2, ff_w2 + (size_t)i * 256 * 64, nullptr,
                                  X1, nullptr, X0, HWPIX);
  }

  k_nhwc2nchw_f32<<<gC, blk, 0, stream>>>(X0, (float*)d_out);
}
